// Memory_attention_3659312136769
// MI455X (gfx1250) — matrix operand order repaired and checked by reading, not compiled, not run on hardware
//
#include <hip/hip_runtime.h>

typedef __attribute__((ext_vector_type(16))) _Float16 v16h;
typedef __attribute__((ext_vector_type(8)))  _Float16 v8h;
typedef __attribute__((ext_vector_type(8)))  float    v8f;

#define DIM   1024
#define HEADS 16
#define HDIM  64
#define BATCH 8
#define N1    1024
#define N2    1024
#define SLOTS 64
#define NKV   (N2 + SLOTS)     // 1088
#define MROWS (BATCH * N1)     // 8192
#define SOFTMAX_SCALE 0.125f   // 1/sqrt(64)

// ---------------------------------------------------------------------------
// WMMA helper: D(16x16 f32) = A(16x32 f16) * B(32x16 f16) + C
// ---------------------------------------------------------------------------
static __device__ __forceinline__ v8f wmma16x16x32(v16h a, v16h b, v8f c) {
  return __builtin_amdgcn_wmma_f32_16x16x32_f16(false, a, false, b,
                                                (short)0, c, false, false);
}

static __device__ __forceinline__ v16h cat16(v8h lo, v8h hi) {
  v16h r;
#pragma unroll
  for (int i = 0; i < 8; ++i) { r[i] = lo[i]; r[i + 8] = hi[i]; }
  return r;
}

// ---------------------------------------------------------------------------
// CDNA5 async copy: per-lane 16B global -> LDS, tracked by ASYNCcnt.
// LDS destination is the 32-bit LDS byte offset (generic LDS addr truncates
// to addr[31:0] per ISA flat addressing).
// ---------------------------------------------------------------------------
static __device__ __forceinline__ void async_ld_b128(unsigned lds_off,
                                                     const void* gaddr) {
  asm volatile("global_load_async_to_lds_b128 %0, %1, off"
               :: "v"(lds_off), "v"(gaddr) : "memory");
}
static __device__ __forceinline__ void wait_async_le4() {
  asm volatile("s_wait_asynccnt 0x4" ::: "memory");
}
static __device__ __forceinline__ void wait_async_0() {
  asm volatile("s_wait_asynccnt 0x0" ::: "memory");
}
static __device__ __forceinline__ unsigned lds_off_of(const void* p) {
  return (unsigned)(uintptr_t)p;
}

// ---------------------------------------------------------------------------
// Elementwise fp32 -> fp16 convert
// ---------------------------------------------------------------------------
__global__ void cvt_f32_f16(const float* __restrict__ in,
                            _Float16* __restrict__ out, int n) {
  int i = blockIdx.x * blockDim.x + threadIdx.x;
  int stride = gridDim.x * blockDim.x;
  for (; i < n; i += stride) out[i] = (_Float16)in[i];
}

// W[K,N] fp32 -> Wt[N,K] fp16 (B^T so WMMA B-operand loads are contiguous)
__global__ void cvt_transpose_w(const float* __restrict__ W,
                                _Float16* __restrict__ Wt) {
  int i = blockIdx.x * blockDim.x + threadIdx.x;   // 1M threads
  int k = i >> 10, n = i & 1023;
  Wt[(size_t)n * DIM + k] = (_Float16)W[i];
}

// memory_k [B,SLOTS,DIM] -> Kh[B,H,NKV,64] at kv = N2 + slot
__global__ void cvt_memk(const float* __restrict__ mk,
                         _Float16* __restrict__ Kh) {
  int i = blockIdx.x * blockDim.x + threadIdx.x;   // B*SLOTS*DIM = 524288
  int b = i >> 16;
  int s = (i >> 10) & 63;
  int c = i & 1023;
  int h = c >> 6, d = c & 63;
  Kh[(((size_t)(b * HEADS + h) * NKV) + N2 + s) * HDIM + d] = (_Float16)mk[i];
}

// memory_v [B,SLOTS,DIM] -> Vt[B,H,64,NKV] (transposed) at kv = N2 + slot
__global__ void cvt_memv(const float* __restrict__ mv,
                         _Float16* __restrict__ Vt) {
  int i = blockIdx.x * blockDim.x + threadIdx.x;
  int b = i >> 16;
  int s = (i >> 10) & 63;
  int c = i & 1023;
  int h = c >> 6, d = c & 63;
  Vt[(((size_t)(b * HEADS + h) * HDIM) + d) * NKV + N2 + s] = (_Float16)mv[i];
}

// ---------------------------------------------------------------------------
// WMMA GEMM: out = A[M,1024] @ W + bias, W given as Wt[N,K] (transposed).
// Block = 4 waves = 128x64 tile; each wave 32x64 (8 wmma / 32-wide K step).
// The shared Wt tile [64n x 32k] is double-buffered in LDS via async copies.
// ---------------------------------------------------------------------------
enum { MODE_PLAIN = 0, MODE_QH = 1, MODE_KH = 2, MODE_VT = 3 };

template <typename OutT, int MODE>
__global__ __launch_bounds__(128) void gemm_bias_wmma(
    const _Float16* __restrict__ A,    // [M x 1024] row-major fp16
    const _Float16* __restrict__ Wt,   // [1024 x 1024] = B^T fp16
    const float* __restrict__ bias,    // [1024] fp32
    OutT* __restrict__ out) {
  __shared__ _Float16 wbuf[2][64 * 32];    // 8 KB

  const int tid  = threadIdx.x;
  const int lane = tid & 31;
  const int wave = tid >> 5;
  const int lo = lane & 15;
  const int hi = lane >> 4;
  const int m0 = blockIdx.x * 128 + wave * 32;
  const int n0 = blockIdx.y * 64;

  // stage Wt tile [n0..n0+63] x [k0..k0+31] into wbuf[buf]: 256 x 16B chunks
  auto stageW = [&](int buf, int k0) {
#pragma unroll
    for (int j = 0; j < 2; ++j) {
      const int c = tid + 128 * j;
      const int n = c >> 2, k8 = (c & 3) * 8;
      async_ld_b128(lds_off_of(&wbuf[buf][n * 32 + k8]),
                    Wt + (size_t)(n0 + n) * DIM + k0 + k8);
    }
  };

  v8f c[2][4] = {};

  // A operands: two 16-row tiles; lane=row, elements K = k0 + hi*8 + e / +16
  const _Float16* arow0 = A + (size_t)(m0 + lo) * DIM + hi * 8;
  const _Float16* arow1 = A + (size_t)(m0 + 16 + lo) * DIM + hi * 8;

  stageW(0, 0);
#pragma unroll 1
  for (int k0 = 0; k0 < DIM; k0 += 32) {
    const int cur = (k0 >> 5) & 1;
    const bool more = (k0 + 32) < DIM;
    if (more) stageW(cur ^ 1, k0 + 32);
    if (more) wait_async_le4(); else wait_async_0();
    __syncthreads();

    if (more) __builtin_prefetch(arow0 + k0 + 32, 0, 0);  // global_prefetch_b8
    v16h a0 = cat16(*(const v8h*)(arow0 + k0), *(const v8h*)(arow0 + k0 + 16));
    v16h a1 = cat16(*(const v8h*)(arow1 + k0), *(const v8h*)(arow1 + k0 + 16));

    const _Float16* wb = &wbuf[cur][0];
#pragma unroll
    for (int t = 0; t < 4; ++t) {
      // B operand: lane = col n; elements: K = hi*8 + e / +16
      const _Float16* brow = wb + (t * 16 + lo) * 32 + hi * 8;
      v16h b = cat16(*(const v8h*)brow, *(const v8h*)(brow + 16));
      c[0][t] = wmma16x16x32(a0, b, c[0][t]);
      c[1][t] = wmma16x16x32(a1, b, c[1][t]);
    }
    __syncthreads();
  }

  // Epilogue: C layout = lane(lo)=col, vgpr r = row (lo half: +r, hi: +8+r)
#pragma unroll
  for (int mi = 0; mi < 2; ++mi) {
#pragma unroll
    for (int t = 0; t < 4; ++t) {
      const int n = n0 + t * 16 + lo;
      const float bval = bias[n];
      const int hcol = n >> 6, d = n & 63;
#pragma unroll
      for (int r = 0; r < 8; ++r) {
        const int m = m0 + mi * 16 + r + hi * 8;
        const float val = c[mi][t][r] + bval;
        size_t idx;
        if (MODE == MODE_PLAIN) {
          idx = (size_t)m * DIM + n;
        } else {
          const int bb = m >> 10, r1 = m & 1023;
          if (MODE == MODE_QH)
            idx = ((size_t)(bb * HEADS + hcol) * N1 + r1) * HDIM + d;
          else if (MODE == MODE_KH)
            idx = ((size_t)(bb * HEADS + hcol) * NKV + r1) * HDIM + d;
          else  // MODE_VT: store V transposed [B,H,64,NKV]
            idx = ((size_t)(bb * HEADS + hcol) * HDIM + d) * NKV + r1;
        }
        out[idx] = (OutT)val;
      }
    }
  }
}

// ---------------------------------------------------------------------------
// Flash attention: block = 4 waves = 4 consecutive 16-query tiles of the SAME
// (b,h). K tile [32x64] and V^T tile [64x32] are shared by all 4 waves and
// double-buffered in LDS via async copies. S^T = K*Q^T so softmax stats are
// per-lane scalars; O^T = V^T*P^T with P regs reused as B-operand.
// ---------------------------------------------------------------------------
__global__ __launch_bounds__(128) void attn_wmma(
    const _Float16* __restrict__ Qh,   // [B,H,N1,64]
    const _Float16* __restrict__ Kh,   // [B,H,NKV,64]
    const _Float16* __restrict__ Vt,   // [B,H,64,NKV]
    _Float16* __restrict__ Oh) {       // [B,N1,DIM]
  __shared__ _Float16 kbuf[2][32 * 64];   // 8 KB
  __shared__ _Float16 vbuf[2][64 * 32];   // 8 KB

  const int tid  = threadIdx.x;
  const int lane = tid & 31;
  const int wave = tid >> 5;
  const int lo = lane & 15, hi = lane >> 4;

  const int tileBase = blockIdx.x * 4;        // 4 q-tiles share (b,h)
  const int h = (tileBase >> 6) & 15;
  const int b = tileBase >> 10;
  const int qt = (tileBase & 63) + wave;
  const int q0 = qt * 16;
  const size_t bh = (size_t)(b * HEADS + h);

  const _Float16* kgbase = Kh + bh * NKV * HDIM;
  const _Float16* vgbase = Vt + bh * HDIM * NKV;

  // stage K [32 kv x 64 d] and V^T [64 d x 32 kv]: 256+256 x 16B chunks
  auto stageKV = [&](int buf, int kv0) {
#pragma unroll
    for (int j = 0; j < 2; ++j) {
      const int c = tid + 128 * j;
      const int kvr = c >> 3, d8 = (c & 7) * 8;
      async_ld_b128(lds_off_of(&kbuf[buf][kvr * 64 + d8]),
                    kgbase + (size_t)(kv0 + kvr) * HDIM + d8);
      const int dr = c >> 2, k8 = (c & 3) * 8;
      async_ld_b128(lds_off_of(&vbuf[buf][dr * 32 + k8]),
                    vgbase + (size_t)dr * NKV + kv0 + k8);
    }
  };

  // Q as B-operand (lane = q column, elements: d = hi*8 + e / +16): stays in regs
  const _Float16* qrow = Qh + (bh * N1 + q0 + lo) * HDIM;
  const v16h qb0 = cat16(*(const v8h*)(qrow + hi * 8),
                         *(const v8h*)(qrow + hi * 8 + 16));
  const v16h qb1 = cat16(*(const v8h*)(qrow + 32 + hi * 8),
                         *(const v8h*)(qrow + 32 + hi * 8 + 16));

  float m_i = -1e30f, l_i = 0.0f;
  v8f o[4] = {};

  stageKV(0, 0);
#pragma unroll 1
  for (int it = 0; it < NKV / 32; ++it) {
    const int kv0 = it * 32;
    const int cur = it & 1;
    const bool more = (it + 1) < (NKV / 32);
    if (more) stageKV(cur ^ 1, kv0 + 32);
    if (more) wait_async_le4(); else wait_async_0();
    __syncthreads();

    // K as A-operand from LDS (lane = kv row, elements: d = hi*8 + e / +16)
    const _Float16* kb = &kbuf[cur][0];
    const _Float16* kr0 = kb + lo * 64 + hi * 8;
    const _Float16* kr1 = kb + (16 + lo) * 64 + hi * 8;
    v16h k00 = cat16(*(const v8h*)(kr0),      *(const v8h*)(kr0 + 16));
    v16h k01 = cat16(*(const v8h*)(kr0 + 32), *(const v8h*)(kr0 + 48));
    v16h k10 = cat16(*(const v8h*)(kr1),      *(const v8h*)(kr1 + 16));
    v16h k11 = cat16(*(const v8h*)(kr1 + 32), *(const v8h*)(kr1 + 48));

    v8f s0 = {}, s1 = {};                 // S^T tiles: rows kv, cols q
    s0 = wmma16x16x32(k00, qb0, s0);
    s0 = wmma16x16x32(k01, qb1, s0);
    s1 = wmma16x16x32(k10, qb0, s1);
    s1 = wmma16x16x32(k11, qb1, s1);

    // Online softmax over q-rows (lane-local + one cross-half exchange)
    float mt = -1e30f;
#pragma unroll
    for (int e = 0; e < 8; ++e) {
      s0[e] *= SOFTMAX_SCALE;
      s1[e] *= SOFTMAX_SCALE;
      mt = fmaxf(mt, fmaxf(s0[e], s1[e]));
    }
    mt = fmaxf(mt, __shfl_xor(mt, 16, 32));
    const float m_new = fmaxf(m_i, mt);
    const float alpha = __expf(m_i - m_new);
    float rs = 0.0f;
#pragma unroll
    for (int e = 0; e < 8; ++e) {
      s0[e] = __expf(s0[e] - m_new);
      s1[e] = __expf(s1[e] - m_new);
      rs += s0[e] + s1[e];
    }
    rs += __shfl_xor(rs, 16, 32);
    l_i = l_i * alpha + rs;
    m_i = m_new;
#pragma unroll
    for (int t = 0; t < 4; ++t)
#pragma unroll
      for (int r = 0; r < 8; ++r) o[t][r] *= alpha;

    // Assemble P^T as B-operand: lane = q col, K = kv = hi*16 + e.
    v16h pb;
#pragma unroll
    for (int e = 0; e < 8; ++e) {
      const float x0 = __shfl_xor(s0[e], 16, 32);
      const float x1 = __shfl_xor(s1[e], 16, 32);
      pb[e]     = (_Float16)(hi ? x1 : s0[e]);
      pb[e + 8] = (_Float16)(hi ? s1[e] : x0);
    }

    // O^T += V^T * P^T over 4 d-tiles; V^T rows contiguous in LDS
    const _Float16* vb = &vbuf[cur][0];
#pragma unroll
    for (int t = 0; t < 4; ++t) {
      const _Float16* vr = vb + (t * 16 + lo) * 32 + hi * 16;
      v16h va = cat16(*(const v8h*)vr, *(const v8h*)(vr + 8));
      o[t] = wmma16x16x32(va, pb, o[t]);
    }
    __syncthreads();
  }

  // Normalize and store O (O^T C-layout: lane = q, vgpr r = d)
  const float inv = 1.0f / l_i;
  _Float16* orow = Oh + ((size_t)(b * N1) + q0 + lo) * DIM + h * HDIM;
#pragma unroll
  for (int t = 0; t < 4; ++t) {
    v8h ov;
#pragma unroll
    for (int r = 0; r < 8; ++r) ov[r] = (_Float16)(o[t][r] * inv);
    *(v8h*)(orow + t * 16 + hi * 8) = ov;
  }
}

// ---------------------------------------------------------------------------
extern "C" void kernel_launch(void* const* d_in, const int* in_sizes, int n_in,
                              void* d_out, int out_size, void* d_ws,
                              size_t ws_size, hipStream_t stream) {
  const float* x1 = (const float*)d_in[0];
  const float* x2 = (const float*)d_in[1];
  const float* mk = (const float*)d_in[2];
  const float* mv = (const float*)d_in[3];
  const float* Wq = (const float*)d_in[4];
  const float* bq = (const float*)d_in[5];
  const float* Wk = (const float*)d_in[6];
  const float* bk = (const float*)d_in[7];
  const float* Wv = (const float*)d_in[8];
  const float* bv = (const float*)d_in[9];
  const float* Wp = (const float*)d_in[10];
  const float* bp = (const float*)d_in[11];
  float* out = (float*)d_out;

  char* ws = (char*)d_ws;
  size_t off = 0;
  auto alloc = [&](size_t bytes) -> char* {
    char* p = ws + off;
    off += (bytes + 255) & ~(size_t)255;
    return p;
  };
  _Float16* x1h = (_Float16*)alloc((size_t)MROWS * DIM * 2);
  _Float16* x2h = (_Float16*)alloc((size_t)MROWS * DIM * 2);
  _Float16* Wqt = (_Float16*)alloc((size_t)DIM * DIM * 2);
  _Float16* Wkt = (_Float16*)alloc((size_t)DIM * DIM * 2);
  _Float16* Wvt = (_Float16*)alloc((size_t)DIM * DIM * 2);
  _Float16* Wpt = (_Float16*)alloc((size_t)DIM * DIM * 2);
  _Float16* Qh  = (_Float16*)alloc((size_t)BATCH * HEADS * N1 * HDIM * 2);
  _Float16* Kh  = (_Float16*)alloc((size_t)BATCH * HEADS * NKV * HDIM * 2);
  _Float16* Vt  = (_Float16*)alloc((size_t)BATCH * HEADS * NKV * HDIM * 2);
  _Float16* Oh  = x1h;  // x1h dead after Q projection; reuse for attn output

  // --- converts ---
  cvt_f32_f16<<<2048, 256, 0, stream>>>(x1, x1h, MROWS * DIM);
  cvt_f32_f16<<<2048, 256, 0, stream>>>(x2, x2h, MROWS * DIM);
  cvt_transpose_w<<<4096, 256, 0, stream>>>(Wq, Wqt);
  cvt_transpose_w<<<4096, 256, 0, stream>>>(Wk, Wkt);
  cvt_transpose_w<<<4096, 256, 0, stream>>>(Wv, Wvt);
  cvt_transpose_w<<<4096, 256, 0, stream>>>(Wp, Wpt);
  cvt_memk<<<2048, 256, 0, stream>>>(mk, Kh);
  cvt_memv<<<2048, 256, 0, stream>>>(mv, Vt);

  // --- projections (WMMA GEMMs, async-LDS staged B) ---
  dim3 gg(MROWS / 128, DIM / 64);   // (64, 16)
  gemm_bias_wmma<_Float16, MODE_QH><<<gg, 128, 0, stream>>>(x1h, Wqt, bq, Qh);
  gemm_bias_wmma<_Float16, MODE_KH><<<gg, 128, 0, stream>>>(x2h, Wkt, bk, Kh);
  gemm_bias_wmma<_Float16, MODE_VT><<<gg, 128, 0, stream>>>(x2h, Wvt, bv, Vt);

  // --- flash attention (async-LDS staged K/V) ---
  attn_wmma<<<(BATCH * HEADS * (N1 / 16)) / 4, 128, 0, stream>>>(Qh, Kh, Vt, Oh);

  // --- output projection (fp32 out) ---
  gemm_bias_wmma<float, MODE_PLAIN><<<gg, 128, 0, stream>>>(Oh, Wpt, bp, out);
}
